// HighLevelGuidedFrequencyFusion_4887672783251
// MI455X (gfx1250) — matrix operand order repaired and checked by reading, not compiled, not run on hardware
//
#include <hip/hip_runtime.h>
#include <math.h>

typedef __attribute__((ext_vector_type(16))) _Float16 v16h;
typedef __attribute__((ext_vector_type(8)))  float    v8f;
typedef __attribute__((ext_vector_type(4)))  unsigned tdm_g0_t;
typedef __attribute__((ext_vector_type(8)))  int      tdm_g1_t;
typedef __attribute__((ext_vector_type(4)))  int      tdm_g2_t;

#define DEVI __device__ __forceinline__

#if __has_builtin(__builtin_amdgcn_tensor_load_to_lds) && __has_builtin(__builtin_amdgcn_s_wait_tensorcnt)
#define USE_TDM 1
#else
#define USE_TDM 0
#endif

struct Lin   { const float* w; const float* b; };
struct HeadP { const float *w1,*b1,*g,*bn,*w2,*b2; };
struct ScoreP{ Lin tok, intent, coord, out; };
struct InterP{ Lin q, k, v, o, gate; };
struct BypP  { Lin h, o, gate; };
struct TdP   { Lin f1, f2, a1, a2, r1, r2; };

DEVI float dgelu(float x){ return 0.5f*x*(1.f+tanhf(0.7978845608f*(x+0.044715f*x*x*x))); }

// LDS fragment-major permutation helpers (v_wmma_f32_16x16x32_f16 layouts).
DEVI int a_slot(int r, int c){
  int wv = r >> 4, mrow = r & 15;
  int grp = (c >> 3) & 1;
  int e = (c & 7) + ((c >> 4) << 3);
  return ((wv*32) + (mrow + (grp << 4)))*16 + e;
}
DEVI int b_slot(int r, int c){
  int nt = c >> 4, ncol = c & 15;
  int grp = (r >> 3) & 1;
  int e = (r & 7) + ((r >> 4) << 3);
  return ((nt*32) + (ncol + (grp << 4)))*16 + e;
}

#if USE_TDM
// Issue a TDM DMA of a 2-D f32 tile (tile_y rows x tile_x cols, row stride
// stride_elems) from global into LDS at lds_off.  tensor_dim0/1 are set to the
// REMAINING extents so the TDM zero-fills out-of-bounds exactly like a bounds
// check.  D# layout per CDNA5 ISA 8.3/8.4 (group0: count/lds/global/type=2;
// group1: data_size=4B, dims, tile dims, dim0 stride).  This toolchain's
// builtin takes 6 args: g0(128b), g1(256b), g2(128b), g3(128b), g4(256b), cpol.
DEVI void tdm_load_2d_f32(const float* gptr, unsigned lds_off,
                          unsigned rem_x, unsigned rem_y,
                          unsigned stride_elems, unsigned tile_x, unsigned tile_y)
{
  unsigned long long ga = (unsigned long long)(uintptr_t)gptr;
  tdm_g0_t g0;
  g0[0] = 1u;                                               // count=1 (valid), user mode
  g0[1] = lds_off;                                          // lds_addr (bytes)
  g0[2] = (unsigned)(ga & 0xffffffffu);                     // global_addr[31:0]
  g0[3] = (unsigned)((ga >> 32) & 0x01ffffffu) | (2u << 30);// global_addr[56:32] | type=2
  tdm_g1_t g1;
  g1[0] = (int)(2u << 16);                                  // wg_mask=0, data_size=2 (4B)
  g1[1] = (int)((rem_x & 0xffffu) << 16);                   // tensor_dim0[15:0]
  g1[2] = (int)((rem_x >> 16) | ((rem_y & 0xffffu) << 16)); // td0[31:16] | td1[15:0]
  g1[3] = (int)((rem_y >> 16) | (tile_x << 16));            // td1[31:16] | tile_dim0
  g1[4] = (int)(tile_y & 0xffffu);                          // tile_dim1 | tile_dim2=0
  g1[5] = (int)stride_elems;                                // tensor_dim0_stride[31:0]
  g1[6] = 0;                                                // stride hi | dim1_stride lo
  g1[7] = 0;
  tdm_g2_t gz = {0,0,0,0};
  tdm_g1_t gz8 = {0,0,0,0,0,0,0,0};
  __builtin_amdgcn_tensor_load_to_lds(g0, g1, gz, gz, gz8, 0);
}
#endif

// ---------------------------------------------------------------------------
// Generic batched WMMA GEMM:  C[M,N] = alpha*A[M,K]@B[K,N] (+C)(+bias)(act)
// TDM stages raw f32 tiles into LDS (TENSORcnt), then the block converts and
// permutes into fragment-major f16 LDS; fragments load as two ds_load_b128;
// f32 accumulate via v_wmma_f32_16x16x32_f16.
// Block = 128 thr = 4 waves; tile 64x64; K step 32.
// BIAS: 0 none, 1 per-column, 2 per-row.  ACT: 0 none, 1 gelu, 2 relu.
// ---------------------------------------------------------------------------
template<int ADD, int BIAS, int ACT>
__global__ void wmma_gemm_f32(const float* __restrict__ A, const float* __restrict__ Bm,
                              float* __restrict__ Cm, const float* __restrict__ bias,
                              int M, int N, int K, long sA, long sB, long sC, float alpha)
{
  __shared__ __align__(32) _Float16 As[2048];
  __shared__ __align__(32) _Float16 Bs[2048];
#if USE_TDM
  __shared__ __align__(16) float Araw[2048];
  __shared__ __align__(16) float Braw[2048];
#endif
  const int bz = blockIdx.z;
  A  += (long)bz * sA;  Bm += (long)bz * sB;  Cm += (long)bz * sC;
  const int tM = blockIdx.y * 64, tN = blockIdx.x * 64;
  const int tid = threadIdx.x, wave = tid >> 5, lane = tid & 31;

  v8f zero = {0.f,0.f,0.f,0.f,0.f,0.f,0.f,0.f};
  v8f acc[4] = {zero, zero, zero, zero};

  for (int k0 = 0; k0 < K; k0 += 32) {
#if USE_TDM
    if (tid < 32) {   // wave 0 drives the tensor data mover
      tdm_load_2d_f32(&A[(long)tM*K + k0],
                      (unsigned)(uintptr_t)&Araw[0],
                      (unsigned)(K - k0), (unsigned)(M - tM), (unsigned)K, 32u, 64u);
      tdm_load_2d_f32(&Bm[(long)k0*N + tN],
                      (unsigned)(uintptr_t)&Braw[0],
                      (unsigned)(N - tN), (unsigned)(K - k0), (unsigned)N, 64u, 32u);
      __builtin_amdgcn_s_wait_tensorcnt(0);
    }
    __syncthreads();
    for (int i = tid; i < 2048; i += 128) {
      int r = i >> 5, c = i & 31;
      As[a_slot(r, c)] = (_Float16)Araw[r*32 + c];
    }
    for (int i = tid; i < 2048; i += 128) {
      int r = i >> 6, c = i & 63;
      Bs[b_slot(r, c)] = (_Float16)Braw[r*64 + c];
    }
#else
    for (int i = tid; i < 2048; i += 128) {
      int r = i >> 5, c = i & 31, gr = tM + r, gc = k0 + c;
      float v = (gr < M && gc < K) ? A[(long)gr*K + gc] : 0.f;
      As[a_slot(r, c)] = (_Float16)v;
    }
    for (int i = tid; i < 2048; i += 128) {
      int r = i >> 6, c = i & 63, gr = k0 + r, gc = tN + c;
      float v = (gr < K && gc < N) ? Bm[(long)gr*N + gc] : 0.f;
      Bs[b_slot(r, c)] = (_Float16)v;
    }
#endif
    // speculative prefetch of the next K tile (global_prefetch_b8)
    if (k0 + 32 < K) {
      __builtin_prefetch(&A[(long)(tM + (tid & 63))*K + (k0 + 32)], 0, 1);
      __builtin_prefetch(&Bm[(long)(k0 + 32 + (tid & 31))*N + tN], 0, 1);
    }
    __syncthreads();
    v16h af = *(const v16h*)&As[(wave*32 + lane)*16];
    #pragma unroll
    for (int nt = 0; nt < 4; ++nt) {
      v16h bf = *(const v16h*)&Bs[(nt*32 + lane)*16];
      acc[nt] = __builtin_amdgcn_wmma_f32_16x16x32_f16(false, af, false, bf,
                                                       (short)0, acc[nt], false, false);
    }
    __syncthreads();
  }
  #pragma unroll
  for (int nt = 0; nt < 4; ++nt) {
    int ncol = tN + 16*nt + (lane & 15);
    #pragma unroll
    for (int r = 0; r < 8; ++r) {
      int mrow = tM + 16*wave + r + ((lane < 16) ? 0 : 8);
      if (mrow < M && ncol < N) {
        float v = alpha * acc[nt][r];
        if (ADD)  v += Cm[(long)mrow*N + ncol];
        if (BIAS == 1) v += bias[ncol];
        if (BIAS == 2) v += bias[mrow];
        if (ACT == 1) v = dgelu(v);
        if (ACT == 2) v = fmaxf(v, 0.f);
        Cm[(long)mrow*N + ncol] = v;
      }
    }
  }
}

static void gemm(hipStream_t s, const float* A, const float* B, float* C, const float* bias,
                 int M, int N, int K, long sA, long sB, long sC, int batch,
                 float alpha, int add, int biasF, int act)
{
  dim3 g((N+63)/64, (M+63)/64, batch), blk(128,1,1);
#define GC(AA,BB,CC) if(add==AA&&biasF==BB&&act==CC){wmma_gemm_f32<AA,BB,CC><<<g,blk,0,s>>>(A,B,C,bias,M,N,K,sA,sB,sC,alpha);return;}
  GC(0,0,0) GC(0,0,1) GC(0,1,0) GC(0,1,1) GC(0,2,0) GC(0,2,1)
  GC(1,0,0) GC(1,0,1) GC(1,1,0) GC(1,1,1) GC(1,2,0) GC(1,2,1)
#undef GC
}

// ---------------------------------------------------------------------------
// Implicit-GEMM 3x3 SAME conv, 64->64 channels, on the WMMA pipe.
// A = weights [64 x 576] (conv weight [cout,cin,3,3] is already row-major),
// B = on-the-fly im2col [576 x H*W] with zero padding, per-batch blockIdx.z.
// ---------------------------------------------------------------------------
template<int ACT, int ADD>
__global__ void wmma_conv3(const float* __restrict__ in, const float* __restrict__ w,
                           const float* __restrict__ bias, float* __restrict__ out,
                           int Hh, int Ww)
{
  __shared__ __align__(32) _Float16 As[2048];
  __shared__ __align__(32) _Float16 Bs[2048];
  const int b = blockIdx.z;
  const long tN = (long)blockIdx.x * 64;
  const long hw = (long)Hh * Ww;
  const int tid = threadIdx.x, wave = tid >> 5, lane = tid & 31;

  v8f zero = {0.f,0.f,0.f,0.f,0.f,0.f,0.f,0.f};
  v8f acc[4] = {zero, zero, zero, zero};

  for (int k0 = 0; k0 < 576; k0 += 32) {
    for (int i = tid; i < 2048; i += 128) {
      int r = i >> 5, c = i & 31, gk = k0 + c;
      As[a_slot(r, c)] = (_Float16)w[(long)r*576 + gk];
    }
    for (int i = tid; i < 2048; i += 128) {
      int r = i >> 6, c = i & 63, gk = k0 + r;
      long n = tN + c;
      float v = 0.f;
      if (n < hw) {
        int ci = gk / 9, jj = gk - ci*9;
        int dy = jj/3 - 1, dx = jj - (jj/3)*3 - 1;
        int y = (int)(n / Ww) + dy, x = (int)(n % Ww) + dx;
        if (y >= 0 && y < Hh && x >= 0 && x < Ww)
          v = in[((long)b*64 + ci)*hw + (long)y*Ww + x];
      }
      Bs[b_slot(r, c)] = (_Float16)v;
    }
    __syncthreads();
    v16h af = *(const v16h*)&As[(wave*32 + lane)*16];
    #pragma unroll
    for (int nt = 0; nt < 4; ++nt) {
      v16h bf = *(const v16h*)&Bs[(nt*32 + lane)*16];
      acc[nt] = __builtin_amdgcn_wmma_f32_16x16x32_f16(false, af, false, bf,
                                                       (short)0, acc[nt], false, false);
    }
    __syncthreads();
  }
  #pragma unroll
  for (int nt = 0; nt < 4; ++nt) {
    long ncol = tN + 16*nt + (lane & 15);
    if (ncol >= hw) continue;
    #pragma unroll
    for (int r = 0; r < 8; ++r) {
      int co = 16*wave + r + ((lane < 16) ? 0 : 8);
      float v = acc[nt][r] + bias[co];
      if (ACT == 1) v = dgelu(v);
      if (ACT == 2) v = fmaxf(v, 0.f);
      long oi = ((long)b*64 + co)*hw + ncol;
      if (ADD) v += out[oi];
      out[oi] = v;
    }
  }
}

static void conv3x3(hipStream_t s, const float* in, const Lin& p, float* out,
                    int Hh, int Ww, int act, int add)
{
  long hw = (long)Hh*Ww;
  dim3 g((unsigned)((hw + 63)/64), 1, 8), blk(128,1,1);
#define CC(AA,DD) if(act==AA&&add==DD){wmma_conv3<AA,DD><<<g,blk,0,s>>>(in,p.w,p.b,out,Hh,Ww);return;}
  CC(0,0) CC(1,0) CC(2,0) CC(0,1)
#undef CC
}

#define EW(tot) dim3((unsigned)((((long)(tot))+255)/256)), 256

// ------------------------- small kernels -----------------------------------
__global__ void gen_dft(float* cs, float* sn, int N){
  int i = blockIdx.x*256 + threadIdx.x;
  if (i >= N*N) return;
  int r = i / N, c = i % N;
  int m = (int)(((long)r*c) % N);
  float a = 6.2831853071795864f * (float)m / (float)N;
  cs[i] = cosf(a); sn[i] = sinf(a);
}

__global__ void avgpool2_k(const float* in, float* out, int Hi, int Wi, long tot){
  long i = (long)blockIdx.x*256 + threadIdx.x; if (i >= tot) return;
  int Wo = Wi>>1, Ho = Hi>>1;
  int x = (int)(i % Wo); long r = i / Wo; int y = (int)(r % Ho); long bc = r / Ho;
  const float* p = in + bc*(long)Hi*Wi;
  out[i] = 0.25f*(p[(2*y)*Wi+2*x] + p[(2*y)*Wi+2*x+1] + p[(2*y+1)*Wi+2*x] + p[(2*y+1)*Wi+2*x+1]);
}

__global__ void mean_hw(const float* vis, const float* ir, float* pooled, int HW){
  int cc = blockIdx.x % 128, b = blockIdx.x / 128, t = threadIdx.x;
  const float* src = (cc < 64) ? vis + ((long)b*64+cc)*HW : ir + ((long)b*64+(cc-64))*HW;
  __shared__ float red[256];
  float s = 0.f;
  for (int i = t; i < HW; i += 256) s += src[i];
  red[t] = s; __syncthreads();
  for (int st = 128; st; st >>= 1){ if (t < st) red[t] += red[t+st]; __syncthreads(); }
  if (t == 0) pooled[b*128+cc] = red[0] / (float)HW;
}

__global__ void router_heads(const float* pooled, Lin r1, Lin r2, const float* prompts,
                             HeadP h0, HeadP h1, HeadP h2, float* intents){
  __shared__ float hb[64], ti[64], w5[8], tmp[64], tmp2[64], red[64];
  int t = threadIdx.x, b = blockIdx.x;
  float s = r1.b[t];
  for (int i = 0; i < 128; ++i) s += pooled[b*128+i]*r1.w[i*64+t];
  hb[t] = dgelu(s);
  __syncthreads();
  if (t < 5){ float s2 = r2.b[t]; for (int i = 0; i < 64; ++i) s2 += hb[i]*r2.w[i*5+t]; w5[t]=s2; }
  __syncthreads();
  if (t == 0){
    float m = w5[0]; for (int i = 1; i < 5; ++i) m = fmaxf(m, w5[i]);
    float su = 0.f; for (int i = 0; i < 5; ++i){ w5[i] = expf(w5[i]-m); su += w5[i]; }
    for (int i = 0; i < 5; ++i) w5[i] /= su;
  }
  __syncthreads();
  { float s3 = 0.f; for (int p = 0; p < 5; ++p) s3 += w5[p]*prompts[p*64+t]; ti[t]=s3; }
  __syncthreads();
  HeadP hs[3] = {h0, h1, h2};
  for (int l = 0; l < 3; ++l){
    HeadP hp = hs[l];
    float s4 = hp.b1[t]; for (int i = 0; i < 64; ++i) s4 += ti[i]*hp.w1[i*64+t];
    tmp[t] = s4; red[t] = s4; __syncthreads();
    for (int st = 32; st; st >>= 1){ if (t < st) red[t] += red[t+st]; __syncthreads(); }
    float mean = red[0] / 64.f; __syncthreads();
    float d = tmp[t] - mean; red[t] = d*d; __syncthreads();
    for (int st = 32; st; st >>= 1){ if (t < st) red[t] += red[t+st]; __syncthreads(); }
    float var = red[0] / 64.f; __syncthreads();
    tmp2[t] = dgelu(d*rsqrtf(var+1e-5f)*hp.g[t] + hp.bn[t]); __syncthreads();
    float s5 = hp.b2[t]; for (int i = 0; i < 64; ++i) s5 += tmp2[i]*hp.w2[i*64+t];
    intents[(l*8+b)*64+t] = s5;
    __syncthreads();
  }
}

__global__ void amp_phase(const float* re, const float* im, float* am, float* ph, long tot){
  long i = (long)blockIdx.x*256 + threadIdx.x; if (i >= tot) return;
  float r = re[i], m = im[i];
  am[i] = sqrtf(r*r + m*m);
  ph[i] = atan2f(m, r);
}

__global__ void polar_kernel(const float* am, const float* ph, float* re, float* im, long tot){
  long i = (long)blockIdx.x*256 + threadIdx.x; if (i >= tot) return;
  float a = am[i], p = ph[i];
  re[i] = a*cosf(p); im[i] = a*sinf(p);
}

__global__ void patchify_k(const float* x, float* t, int hp, int wp, long tot){
  long i = (long)blockIdx.x*256 + threadIdx.x; if (i >= tot) return;
  int tok = (int)(i & 1023); long r = i >> 10;
  int n = (int)(r % (hp*wp)); int b = (int)(r / (hp*wp));
  int c = tok >> 4, dy = (tok >> 2) & 3, dx = tok & 3;
  int py = n / wp, px = n % wp, Hh = hp*4, Ww = wp*4;
  t[i] = x[(((long)b*64 + c)*Hh + py*4+dy)*Ww + px*4+dx];
}

__global__ void unpatchify_k(const float* t, float* x, int hp, int wp, long tot){
  long i = (long)blockIdx.x*256 + threadIdx.x; if (i >= tot) return;
  int Ww = wp*4, Hh = hp*4;
  int xx = (int)(i % Ww); long r = i / Ww;
  int yy = (int)(r % Hh); r /= Hh;
  int c = (int)(r % 64); int b = (int)(r / 64);
  int py = yy >> 2, dy = yy & 3, px = xx >> 2, dx = xx & 3;
  x[i] = t[((long)b*hp*wp + py*wp + px)*1024 + c*16 + dy*4 + dx];
}

__global__ void coord_embed(const float* w, const float* b, float* out, int hp, int wp){
  long i = (long)blockIdx.x*256 + threadIdx.x;
  if (i >= (long)hp*wp*128) return;
  int j = (int)(i & 127); int n = (int)(i >> 7);
  int py = n / wp, px = n % wp;
  float y = (float)py/(float)(hp-1), x = (float)px/(float)(wp-1);
  out[i] = y*w[j] + x*w[128+j] + b[j];
}

__global__ void add3_gelu(float* h, const float* ih, const float* ch, int n, long tot){
  long i = (long)blockIdx.x*256 + threadIdx.x; if (i >= tot) return;
  int j = (int)(i & 127); long r = i >> 7;
  int nn = (int)(r % n); int b = (int)(r / n);
  h[i] = dgelu(h[i] + ih[b*128+j] + ch[nn*128+j]);
}

__global__ void score_out(const float* h, const float* w, const float* b, float* sc, long rows){
  long i = (long)blockIdx.x*256 + threadIdx.x; if (i >= rows) return;
  const float* hp_ = h + i*128;
  float s = b[0];
  for (int j = 0; j < 128; ++j) s += hp_[j]*w[j];
  sc[i] = s;
}

__global__ void topk_select(const float* score, int n, int k, int* idx){
  int b = blockIdx.x, t = threadIdx.x;
  const float* s = score + (long)b*n;
  __shared__ float rf[256]; __shared__ int ri[256]; __shared__ float bnd[2]; __shared__ int cnt;
  float lo = 3.4e38f, hi = -3.4e38f;
  for (int i = t; i < n; i += 256){ float v = s[i]; lo = fminf(lo, v); hi = fmaxf(hi, v); }
  rf[t] = lo; __syncthreads();
  for (int st = 128; st; st >>= 1){ if (t < st) rf[t] = fminf(rf[t], rf[t+st]); __syncthreads(); }
  if (t == 0) bnd[0] = rf[0]; __syncthreads();
  rf[t] = hi; __syncthreads();
  for (int st = 128; st; st >>= 1){ if (t < st) rf[t] = fmaxf(rf[t], rf[t+st]); __syncthreads(); }
  if (t == 0) bnd[1] = rf[0]; __syncthreads();
  float a = bnd[0], c = bnd[1];
  for (int it = 0; it < 32; ++it){
    float mid = 0.5f*(a+c);
    int cl = 0;
    for (int i = t; i < n; i += 256) cl += (s[i] >= mid) ? 1 : 0;
    ri[t] = cl; __syncthreads();
    for (int st = 128; st; st >>= 1){ if (t < st) ri[t] += ri[t+st]; __syncthreads(); }
    int tot = ri[0]; __syncthreads();
    if (tot >= k) a = mid; else c = mid;
  }
  if (t == 0) cnt = 0; __syncthreads();
  for (int i = t; i < n; i += 256){
    if (s[i] >= a){ int p = atomicAdd(&cnt, 1); if (p < k) idx[(long)b*k+p] = i; }
  }
  __syncthreads();
  if (t == 0){ int have = (cnt < k) ? cnt : k; for (int j = have; j < k; ++j) idx[(long)b*k+j] = j; }
}

__global__ void gather_tok(const float* src, const int* idx, float* dst, int n, int k, long tot){
  long i = (long)blockIdx.x*256 + threadIdx.x; if (i >= tot) return;
  int tk = (int)(i & 1023); long r = i >> 10;
  int j = (int)(r % k); int b = (int)(r / k);
  dst[i] = src[((long)b*n + idx[(long)b*k+j])*1024 + tk];
}

__global__ void weak_combine(const float* wb, const float* gate, const float* vt,
                             const float* irt, float* out, int n, long tot){
  long i = (long)blockIdx.x*256 + threadIdx.x; if (i >= tot) return;
  int t = (int)(i & 1023); int b = (int)((i >> 10) / n);
  out[i] = wb[i]*(1.f + 0.1f*tanhf(gate[b*1024+t])) + 0.5f*(vt[i]+irt[i]);
}

__global__ void strong_scatter(const float* oO, const float* gate, const float* vs,
                               const float* is_, const int* idx, float* ftok,
                               int n, int k, long tot){
  long i = (long)blockIdx.x*256 + threadIdx.x; if (i >= tot) return;
  int t = (int)(i & 1023); long r = i >> 10;
  int j = (int)(r % k); int b = (int)(r / k);
  float v = oO[i]*(1.f + 0.1f*tanhf(gate[b*1024+t])) + 0.5f*(vs[i]+is_[i]);
  ftok[((long)b*n + idx[(long)b*k+j])*1024 + t] = v;
}

__global__ void attention(const float* q, const float* kb, const float* vb, float* o, int kk){
  int qi = blockIdx.x, h = blockIdx.y, b = blockIdx.z, t = threadIdx.x;
  __shared__ float sm[640]; __shared__ float red[256];
  const float* qp = q + ((long)b*kk + qi)*128 + h*32;
  for (int j = t; j < kk; j += 256){
    const float* kp = kb + ((long)b*kk + j)*128 + h*32;
    float s = 0.f;
    for (int d = 0; d < 32; ++d) s += qp[d]*kp[d];
    sm[j] = s * 0.176776695f;   // 1/sqrt(32)
  }
  __syncthreads();
  float m = -3.4e38f;
  for (int j = t; j < kk; j += 256) m = fmaxf(m, sm[j]);
  red[t] = m; __syncthreads();
  for (int st = 128; st; st >>= 1){ if (t < st) red[t] = fmaxf(red[t], red[t+st]); __syncthreads(); }
  m = red[0]; __syncthreads();
  float ss = 0.f;
  for (int j = t; j < kk; j += 256){ float e = expf(sm[j]-m); sm[j] = e; ss += e; }
  red[t] = ss; __syncthreads();
  for (int st = 128; st; st >>= 1){ if (t < st) red[t] += red[t+st]; __syncthreads(); }
  float inv = 1.f / red[0];
  __syncthreads();
  if (t < 32){
    float acc = 0.f;
    for (int j = 0; j < kk; ++j) acc += sm[j]*vb[((long)b*kk+j)*128 + h*32 + t];
    o[((long)b*kk+qi)*128 + h*32 + t] = acc*inv;
  }
}

__global__ void affine_res(float* x, const float* gb, const float* r1, const float* r2,
                           int Hh, int Ww, long tot){
  long i = (long)blockIdx.x*256 + threadIdx.x; if (i >= tot) return;
  long hw = (long)Hh*Ww;
  int c = (int)((i / hw) % 64); int b = (int)(i / (hw*64));
  float v = x[i]*(1.f + 0.1f*tanhf(gb[b*128+c])) + 0.1f*gb[b*128+64+c];
  if (r1) v += 0.5f*(r1[i]+r2[i]);
  x[i] = v;
}

__global__ void upsample_bilinear(const float* in, float* out, int Hi, int Wi,
                                  int Ho, int Wo, long tot){
  long i = (long)blockIdx.x*256 + threadIdx.x; if (i >= tot) return;
  int x = (int)(i % Wo); long r = i / Wo;
  int y = (int)(r % Ho); long bc = r / Ho;
  float sy = (float)Hi/Ho, sx = (float)Wi/Wo;
  float fy = (y+0.5f)*sy - 0.5f, fx = (x+0.5f)*sx - 0.5f;
  int y0 = (int)floorf(fy), x0 = (int)floorf(fx);
  float wy = fy - y0, wx = fx - x0;
  int y1 = y0+1, x1 = x0+1;
  y0 = y0 < 0 ? 0 : (y0 >= Hi ? Hi-1 : y0);
  y1 = y1 < 0 ? 0 : (y1 >= Hi ? Hi-1 : y1);
  x0 = x0 < 0 ? 0 : (x0 >= Wi ? Wi-1 : x0);
  x1 = x1 < 0 ? 0 : (x1 >= Wi ? Wi-1 : x1);
  const float* p = in + bc*(long)Hi*Wi;
  out[i] = (1.f-wy)*((1.f-wx)*p[(long)y0*Wi+x0] + wx*p[(long)y0*Wi+x1])
         +      wy *((1.f-wx)*p[(long)y1*Wi+x0] + wx*p[(long)y1*Wi+x1]);
}

__global__ void concat_ch(const float* a, const float* b, float* out, long hw, long tot){
  long i = (long)blockIdx.x*256 + threadIdx.x; if (i >= tot) return;
  long pix = i % hw; long r = i / hw;
  int c = (int)(r % 128); int bb = (int)(r / 128);
  out[i] = (c < 64) ? a[((long)bb*64+c)*hw+pix] : b[((long)bb*64+(c-64))*hw+pix];
}

// ---------------------------------------------------------------------------
extern "C" void kernel_launch(void* const* d_in, const int* in_sizes, int n_in,
                              void* d_out, int out_size, void* d_ws, size_t ws_size,
                              hipStream_t stream)
{
  const float* vis = (const float*)d_in[0];
  const float* ir  = (const float*)d_in[1];
  int pi = 2;
  auto nf = [&](){ return (const float*)d_in[pi++]; };
  auto nl = [&](){ Lin l; l.w = nf(); l.b = nf(); return l; };

  const float* prompts = nf();
  Lin router1 = nl(), router2 = nl();
  HeadP heads[3];
  for (int l = 0; l < 3; ++l){ Lin a = nl(), b = nl(), c = nl();
    heads[l].w1=a.w; heads[l].b1=a.b; heads[l].g=b.w; heads[l].bn=b.b; heads[l].w2=c.w; heads[l].b2=c.b; }
  ScoreP asc[3], psc[3];
  for (int l = 0; l < 3; ++l) asc[l] = { nl(), nl(), nl(), nl() };
  for (int l = 0; l < 3; ++l) psc[l] = { nl(), nl(), nl(), nl() };
  InterP ain[3], pin2[3];
  for (int l = 0; l < 3; ++l) ain[l]  = { nl(), nl(), nl(), nl(), nl() };
  for (int l = 0; l < 3; ++l) pin2[l] = { nl(), nl(), nl(), nl(), nl() };
  BypP abyp[3], pbyp[3];
  for (int l = 0; l < 3; ++l) abyp[l] = { nl(), nl(), nl() };
  for (int l = 0; l < 3; ++l) pbyp[l] = { nl(), nl(), nl() };
  Lin lref1[3], lref2[3], laff1[3], laff2[3];
  for (int l = 0; l < 3; ++l) lref1[l] = nl();
  for (int l = 0; l < 3; ++l) lref2[l] = nl();
  for (int l = 0; l < 3; ++l) laff1[l] = nl();
  for (int l = 0; l < 3; ++l) laff2[l] = nl();
  TdP td2 = { nl(), nl(), nl(), nl(), nl(), nl() };
  TdP td1 = { nl(), nl(), nl(), nl(), nl(), nl() };
  Lin fin1 = nl(), fin2 = nl();

  // ---- workspace bump allocator ----
  char* wsb = (char*)d_ws; size_t off = 0;
  auto alloc = [&](size_t elems)->float*{
    float* p = (float*)(wsb + off);
    off += ((elems*sizeof(float) + 255) & ~(size_t)255);
    return p;
  };

  const int HlA[3] = {192, 96, 48};
  float *cs[3], *sn[3];
  for (int l = 0; l < 3; ++l){ cs[l] = alloc((size_t)HlA[l]*HlA[l]); sn[l] = alloc((size_t)HlA[l]*HlA[l]); }
  float* vl1 = alloc((size_t)8*64*96*96);  float* vl2 = alloc((size_t)8*64*48*48);
  float* il1 = alloc((size_t)8*64*96*96);  float* il2 = alloc((size_t)8*64*48*48);
  float* pooled  = alloc(8*128);
  float* intents = alloc(3*8*64);
  float* g1 = alloc(8*128);  float* gb = alloc(8*128);
  const long FSZ = (long)512*192*192;                 // = B*C*H*W at level 0
  float* Fre = alloc(FSZ); float* Fim = alloc(FSZ);
  float* Gre = alloc(FSZ); float* Gim = alloc(FSZ);
  float* ampA = alloc(FSZ); float* phA = alloc(FSZ);
  float* ampB = alloc(FSZ); float* phB = alloc(FSZ);
  float* faMap = alloc(FSZ); float* fpMap = alloc(FSZ);
  const long RMAX = (long)8*2304, KRMAX = (long)8*576;
  float* vt   = alloc(RMAX*1024); float* irt  = alloc(RMAX*1024);
  float* hbuf = alloc(RMAX*1024); float* wbuf = alloc(RMAX*1024);
  float* ftok = alloc(RMAX*1024);
  float* h128 = alloc(RMAX*128);  float* coordh = alloc((size_t)2304*128);
  float* intH = alloc(8*128);     float* scoreb = alloc(RMAX);
  int*   idxb = (int*)alloc(8*576);
  float* vsel = alloc(KRMAX*1024); float* isel = alloc(KRMAX*1024);
  float* qb = alloc(KRMAX*128); float* kb2 = alloc(KRMAX*128);
  float* vb2 = alloc(KRMAX*128); float* ob = alloc(KRMAX*128);
  float* oO = alloc(KRMAX*1024);
  float* gA = alloc(8*1024); float* gB2 = alloc(8*1024);
  float* t1   = alloc(FSZ);
  float* upb  = alloc(FSZ);
  float* catb = alloc((size_t)8*128*192*192);
  float* xTD  = alloc((size_t)8*64*96*96);
  float* fusedL[3];
  fusedL[0] = alloc(FSZ); fusedL[1] = alloc((size_t)8*64*96*96); fusedL[2] = alloc((size_t)8*64*48*48);

  // ---- DFT matrices, pyramid, router ----
  for (int l = 0; l < 3; ++l){
    int N = HlA[l];
    gen_dft<<<EW((long)N*N), 0, stream>>>(cs[l], sn[l], N);
  }
  avgpool2_k<<<EW((long)8*64*96*96), 0, stream>>>(vis, vl1, 192, 192, (long)8*64*96*96);
  avgpool2_k<<<EW((long)8*64*48*48), 0, stream>>>(vl1, vl2, 96, 96, (long)8*64*48*48);
  avgpool2_k<<<EW((long)8*64*96*96), 0, stream>>>(ir,  il1, 192, 192, (long)8*64*96*96);
  avgpool2_k<<<EW((long)8*64*48*48), 0, stream>>>(il1, il2, 96, 96, (long)8*64*48*48);
  mean_hw<<<dim3(8*128), 256, 0, stream>>>(vis, ir, pooled, 192*192);
  router_heads<<<dim3(8), 64, 0, stream>>>(pooled, router1, router2, prompts,
                                           heads[0], heads[1], heads[2], intents);

  // ---- fuse_branch (host lambda) ----
  auto fuse_branch = [&](int Nside, const float* Amap, const float* Imap, const float* fint,
                         const ScoreP& sp, const InterP& ipp, const BypP& bpp, float* outMap){
    int hp = Nside/4, n = hp*hp, k = n/4;
    long rows = (long)8*n, krows = (long)8*k;
    patchify_k<<<EW(rows*1024), 0, stream>>>(Amap, vt, hp, hp, rows*1024);
    patchify_k<<<EW(rows*1024), 0, stream>>>(Imap, irt, hp, hp, rows*1024);
    gemm(stream, vt,  sp.tok.w,            h128, nullptr,  (int)rows, 128, 1024, 0,0,0, 1, 1.f, 0,0,0);
    gemm(stream, irt, sp.tok.w+1024*128,   h128, sp.tok.b, (int)rows, 128, 1024, 0,0,0, 1, 1.f, 1,1,0);
    gemm(stream, fint, sp.intent.w, intH, sp.intent.b, 8, 128, 64, 0,0,0, 1, 1.f, 0,1,0);
    coord_embed<<<EW((long)n*128), 0, stream>>>(sp.coord.w, sp.coord.b, coordh, hp, hp);
    add3_gelu<<<EW(rows*128), 0, stream>>>(h128, intH, coordh, n, rows*128);
    score_out<<<EW(rows), 0, stream>>>(h128, sp.out.w, sp.out.b, scoreb, rows);
    topk_select<<<dim3(8), 256, 0, stream>>>(scoreb, n, k, idxb);
    // weak path
    gemm(stream, vt,  bpp.h.w,                    hbuf, nullptr, (int)rows, 1024, 1024, 0,0,0, 1, 1.f, 0,0,0);
    gemm(stream, irt, bpp.h.w+(long)1024*1024,    hbuf, bpp.h.b, (int)rows, 1024, 1024, 0,0,0, 1, 1.f, 1,1,1);
    gemm(stream, hbuf, bpp.o.w, wbuf, bpp.o.b, (int)rows, 1024, 1024, 0,0,0, 1, 1.f, 0,1,0);
    gemm(stream, fint, bpp.gate.w, gB2, bpp.gate.b, 8, 1024, 64, 0,0,0, 1, 1.f, 0,1,0);
    weak_combine<<<EW(rows*1024), 0, stream>>>(wbuf, gB2, vt, irt, ftok, n, rows*1024);
    // strong path
    gather_tok<<<EW(krows*1024), 0, stream>>>(vt, idxb, vsel, n, k, krows*1024);
    gather_tok<<<EW(krows*1024), 0, stream>>>(irt, idxb, isel, n, k, krows*1024);
    gemm(stream, vsel, ipp.q.w, qb,  ipp.q.b, (int)krows, 128, 1024, 0,0,0, 1, 1.f, 0,1,0);
    gemm(stream, isel, ipp.k.w, kb2, ipp.k.b, (int)krows, 128, 1024, 0,0,0, 1, 1.f, 0,1,0);
    gemm(stream, isel, ipp.v.w, vb2, ipp.v.b, (int)krows, 128, 1024, 0,0,0, 1, 1.f, 0,1,0);
    attention<<<dim3(k, 4, 8), 256, 0, stream>>>(qb, kb2, vb2, ob, k);
    gemm(stream, ob, ipp.o.w, oO, ipp.o.b, (int)krows, 1024, 128, 0,0,0, 1, 1.f, 0,1,0);
    gemm(stream, fint, ipp.gate.w, gA, ipp.gate.b, 8, 1024, 64, 0,0,0, 1, 1.f, 0,1,0);
    strong_scatter<<<EW(krows*1024), 0, stream>>>(oO, gA, vsel, isel, idxb, ftok, n, k, krows*1024);
    unpatchify_k<<<EW((long)8*64*Nside*Nside), 0, stream>>>(ftok, outMap, hp, hp, (long)8*64*Nside*Nside);
  };

  // ---- fuse_level (host lambda): DFT fwd (WMMA GEMMs), branches, DFT inv, refine ----
  auto fuse_level = [&](int l, const float* V, const float* I){
    int N = HlA[l]; long NN = (long)N*N; int nb = 512;
    float inv = 1.f / ((float)N*(float)N);
    const float* Cs = cs[l]; const float* Sn = sn[l];
    const float* fint = intents + l*8*64;
    long tot = NN*nb;
    // forward FFT of vis: Z = F X F   (F = Cs - i Sn; Cs/Sn symmetric)
    gemm(stream, V,  Cs, Gre, nullptr, N,N,N, NN,0,NN, nb,  1.f, 0,0,0);
    gemm(stream, V,  Sn, Gim, nullptr, N,N,N, NN,0,NN, nb, -1.f, 0,0,0);
    gemm(stream, Cs, Gre, Fre, nullptr, N,N,N, 0,NN,NN, nb,  1.f, 0,0,0);
    gemm(stream, Sn, Gim, Fre, nullptr, N,N,N, 0,NN,NN, nb,  1.f, 1,0,0);
    gemm(stream, Cs, Gim, Fim, nullptr, N,N,N, 0,NN,NN, nb,  1.f, 0,0,0);
    gemm(stream, Sn, Gre, Fim, nullptr, N,N,N, 0,NN,NN, nb, -1.f, 1,0,0);
    amp_phase<<<EW(tot), 0, stream>>>(Fre, Fim, ampA, phA, tot);
    // forward FFT of ir
    gemm(stream, I,  Cs, Gre, nullptr, N,N,N, NN,0,NN, nb,  1.f, 0,0,0);
    gemm(stream, I,  Sn, Gim, nullptr, N,N,N, NN,0,NN, nb, -1.f, 0,0,0);
    gemm(stream, Cs, Gre, Fre, nullptr, N,N,N, 0,NN,NN, nb,  1.f, 0,0,0);
    gemm(stream, Sn, Gim, Fre, nullptr, N,N,N, 0,NN,NN, nb,  1.f, 1,0,0);
    gemm(stream, Cs, Gim, Fim, nullptr, N,N,N, 0,NN,NN, nb,  1.f, 0,0,0);
    gemm(stream, Sn, Gre, Fim, nullptr, N,N,N, 0,NN,NN, nb, -1.f, 1,0,0);
    amp_phase<<<EW(tot), 0, stream>>>(Fre, Fim, ampB, phB, tot);
    // routed fusion of amplitude and phase
    fuse_branch(N, ampA, ampB, fint, asc[l], ain[l], abyp[l], faMap);
    fuse_branch(N, phA,  phB,  fint, psc[l], pin2[l], pbyp[l], fpMap);
    polar_kernel<<<EW(tot), 0, stream>>>(faMap, fpMap, Fre, Fim, tot);
    // inverse FFT real part: S = Re(conj(F) Z conj(F)) / N^2
    gemm(stream, Fre, Cs, Gre, nullptr, N,N,N, NN,0,NN, nb,  1.f, 0,0,0);
    gemm(stream, Fim, Sn, Gre, nullptr, N,N,N, NN,0,NN, nb, -1.f, 1,0,0);
    gemm(stream, Fre, Sn, Gim, nullptr, N,N,N, NN,0,NN, nb,  1.f, 0,0,0);
    gemm(stream, Fim, Cs, Gim, nullptr, N,N,N, NN,0,NN, nb,  1.f, 1,0,0);
    gemm(stream, Cs, Gre, ampA, nullptr, N,N,N, 0,NN,NN, nb,  inv, 0,0,0);
    gemm(stream, Sn, Gim, ampA, nullptr, N,N,N, 0,NN,NN, nb, -inv, 1,0,0);
    // refinement convs (implicit-GEMM WMMA) + intent affine + residual
    long cvt = (long)8*64*N*N;
    conv3x3(stream, ampA, lref1[l], t1, N, N, /*act=*/2, /*add=*/0);
    conv3x3(stream, t1, lref2[l], fusedL[l], N, N, 0, 0);
    gemm(stream, fint, laff1[l].w, g1, laff1[l].b, 8, 128, 64, 0,0,0, 1, 1.f, 0,1,1);
    gemm(stream, g1,   laff2[l].w, gb, laff2[l].b, 8, 128, 128, 0,0,0, 1, 1.f, 0,1,0);
    affine_res<<<EW(cvt), 0, stream>>>(fusedL[l], gb, V, I, N, N, cvt);
  };

  fuse_level(0, vis, ir);
  fuse_level(1, vl1, il1);
  fuse_level(2, vl2, il2);

  // ---- top-down decoder ----
  auto topdown = [&](const float* high, int Hhi, const float* skip, int Hsk,
                     const float* fintL, const TdP& tp, float* xout){
    long hw = (long)Hsk*Hsk;
    long tot = (long)8*64*hw;
    upsample_bilinear<<<EW(tot), 0, stream>>>(high, upb, Hhi, Hhi, Hsk, Hsk, tot);
    concat_ch<<<EW((long)8*128*hw), 0, stream>>>(upb, skip, catb, hw, (long)8*128*hw);
    // fuse1: 1x1 conv as batched WMMA GEMM (row bias + gelu)
    gemm(stream, tp.f1.w, catb, t1, tp.f1.b, 64, (int)hw, 128,
         0, (long)128*hw, (long)64*hw, 8, 1.f, 0,2,1);
    conv3x3(stream, t1, tp.f2, xout, Hsk, Hsk, 0, 0);
    gemm(stream, fintL, tp.a1.w, g1, tp.a1.b, 8, 128, 64, 0,0,0, 1, 1.f, 0,1,1);
    gemm(stream, g1,    tp.a2.w, gb, tp.a2.b, 8, 128, 128, 0,0,0, 1, 1.f, 0,1,0);
    affine_res<<<EW(tot), 0, stream>>>(xout, gb, nullptr, nullptr, Hsk, Hsk, tot);
    conv3x3(stream, xout, tp.r1, t1, Hsk, Hsk, 1, 0);
    conv3x3(stream, t1, tp.r2, xout, Hsk, Hsk, 0, 1);
  };

  float* out = (float*)d_out;
  topdown(fusedL[2], 48, fusedL[1], 96, intents + 1*8*64, td2, xTD);
  topdown(xTD, 96, fusedL[0], 192, intents + 0*8*64, td1, out);

  // ---- final refinement: out += conv(gelu(conv(out))) ----
  conv3x3(stream, out, fin1, t1, 192, 192, 1, 0);
  conv3x3(stream, t1, fin2, out, 192, 192, 0, 1);
}
